// DSGCNEncoder_10823317586650
// MI455X (gfx1250) — compile-verified, matrix operand order checked by reading, not run on hardware
//
#include <hip/hip_runtime.h>
#include <hip/hip_bf16.h>
#include <math.h>

// ---------------------------------------------------------------------------
// DSGCN encoder forward for MI455X (gfx1250, wave32, WMMA).
// GEMMs: bf16 operands staged in workspace, v_wmma_f32_16x16x32_bf16,
// 64x16 output strip per wave (4 WMMAs share one B fragment per K-step),
// guard-free K loop with b128 loads.
// ---------------------------------------------------------------------------

#define B_ 64
#define T_ 32
#define N_ 47
#define CIN_ 16
#define D_ 384
#define ROWS_ (B_*T_*N_)    // 96256 = 64*16*94 (multiple of 64)
#define TOK_  (B_*T_)       // 2048
#define NGEO_ 76
#define KGEO_ 480           // 384+76=460 padded to multiple of 32

typedef __attribute__((ext_vector_type(16))) __bf16 v16bf;
typedef __attribute__((ext_vector_type(8)))  __bf16 v8bf;   // 16 bytes
typedef __attribute__((ext_vector_type(4)))  __bf16 v4bf;   // 8 bytes
typedef __attribute__((ext_vector_type(8)))  float  v8f;

__device__ __forceinline__ float gelu_f(float x) {
    return 0.5f * x * (1.0f + erff(x * 0.7071067811865476f));
}

__device__ __forceinline__ __bf16 f2bf(float x) {
    union { float f; unsigned u; } un; un.f = x;
    unsigned r = un.u + 0x7FFFu + ((un.u >> 16) & 1u);   // round to nearest even
    union { unsigned short s; __bf16 b; } o; o.s = (unsigned short)(r >> 16);
    return o.b;
}

// ---------------------------------------------------------------------------
// WMMA GEMM: C[M,N] = act(A[M,K] @ W[N,K]^T + bias) + res
// A, W pre-converted bf16. One wave computes a 64x16 strip (4 M-tiles),
// sharing the B fragment across 4 WMMAs. K % 32 == 0, M % 64 == 0, N % 16 == 0.
// ---------------------------------------------------------------------------
__global__ void k_gemm(const __bf16* __restrict__ A, const __bf16* __restrict__ W,
                       const float* __restrict__ bias, const float* __restrict__ res,
                       float* __restrict__ C, __bf16* __restrict__ Cbf,
                       int M, int N, int K, int act)
{
    int wave = threadIdx.x >> 5;
    int lane = threadIdx.x & 31;
    int tilesN = N >> 4;
    int ngroups = (M >> 6) * tilesN;
    int grp = blockIdx.x * 4 + wave;
    if (grp >= ngroups) return;
    int gm = grp / tilesN, tn = grp % tilesN;
    int half = lane >> 4;          // which K-half of the fragment this lane holds
    int i16  = lane & 15;          // M index (A frag) / N index (B frag)

    // 16-bit fragment layout: lane(half,i16) holds K = khalf*8 + [0..7] and
    // 16 + khalf*8 + [0..7]  ->  two contiguous 16B chunks per operand.
    const __bf16* Wrow = W + (size_t)(tn * 16 + i16) * K + half * 8;
    const __bf16* Ar0 = A + (size_t)(gm * 64 +  0 + i16) * K + half * 8;
    const __bf16* Ar1 = A + (size_t)(gm * 64 + 16 + i16) * K + half * 8;
    const __bf16* Ar2 = A + (size_t)(gm * 64 + 32 + i16) * K + half * 8;
    const __bf16* Ar3 = A + (size_t)(gm * 64 + 48 + i16) * K + half * 8;

    v8f acc0 = {}, acc1 = {}, acc2 = {}, acc3 = {};
    union Frag { v8bf h[2]; v16bf v; };
    for (int kb = 0; kb < K; kb += 32) {
        if (kb + 32 < K) {
            __builtin_prefetch(Wrow + kb + 32, 0, 1);   // global_prefetch_b8
            __builtin_prefetch(Ar0  + kb + 32, 0, 1);
        }
        Frag b, a0, a1, a2, a3;
        b.h[0]  = *(const v8bf*)(Wrow + kb); b.h[1]  = *(const v8bf*)(Wrow + kb + 16);
        a0.h[0] = *(const v8bf*)(Ar0  + kb); a0.h[1] = *(const v8bf*)(Ar0  + kb + 16);
        a1.h[0] = *(const v8bf*)(Ar1  + kb); a1.h[1] = *(const v8bf*)(Ar1  + kb + 16);
        a2.h[0] = *(const v8bf*)(Ar2  + kb); a2.h[1] = *(const v8bf*)(Ar2  + kb + 16);
        a3.h[0] = *(const v8bf*)(Ar3  + kb); a3.h[1] = *(const v8bf*)(Ar3  + kb + 16);
        acc0 = __builtin_amdgcn_wmma_f32_16x16x32_bf16(false, a0.v, false, b.v, (short)0, acc0, false, false);
        acc1 = __builtin_amdgcn_wmma_f32_16x16x32_bf16(false, a1.v, false, b.v, (short)0, acc1, false, false);
        acc2 = __builtin_amdgcn_wmma_f32_16x16x32_bf16(false, a2.v, false, b.v, (short)0, acc2, false, false);
        acc3 = __builtin_amdgcn_wmma_f32_16x16x32_bf16(false, a3.v, false, b.v, (short)0, acc3, false, false);
    }

    // C/D layout: lane holds N=i16; VGPR v holds M = half*8 + v
    int n = tn * 16 + i16;
    float bz = bias ? bias[n] : 0.f;
    v8f accs[4] = {acc0, acc1, acc2, acc3};
#pragma unroll
    for (int s = 0; s < 4; ++s) {
#pragma unroll
        for (int v = 0; v < 8; ++v) {
            int m = gm * 64 + s * 16 + half * 8 + v;
            float y = accs[s][v] + bz;
            if (act) y = gelu_f(y);
            size_t off = (size_t)m * N + n;
            if (res) y += res[off];
            if (C)   C[off] = y;
            if (Cbf) Cbf[off] = f2bf(y);
        }
    }
}

// ---------------------------------------------------------------------------
// f32 -> bf16 conversion (n must be a multiple of 4)
// ---------------------------------------------------------------------------
__global__ void k_tobf(const float* __restrict__ in, __bf16* __restrict__ out, long n4)
{
    long id = (long)blockIdx.x * 256 + threadIdx.x;
    if (id >= n4) return;
    float4 v = ((const float4*)in)[id];
    v4bf o;
    o[0] = f2bf(v.x); o[1] = f2bf(v.y); o[2] = f2bf(v.z); o[3] = f2bf(v.w);
    ((v4bf*)out)[id] = o;
}

// geo_proj weight [384,460] -> bf16 padded to [384,480]
__global__ void k_padw(const float* __restrict__ in, __bf16* __restrict__ out)
{
    int id = blockIdx.x * 256 + threadIdx.x;
    if (id >= D_ * KGEO_) return;
    int k = id % KGEO_, n = id / KGEO_;
    out[id] = (k < D_ + NGEO_) ? f2bf(in[n * (D_ + NGEO_) + k]) : f2bf(0.f);
}

// ---------------------------------------------------------------------------
// Fused: input LN(16) -> proj(16->96)+bias -> LN(96) -> GELU -> face mask
// ---------------------------------------------------------------------------
__global__ void k_in_proj(const float* __restrict__ x,
                          const float* __restrict__ inw, const float* __restrict__ inb,
                          const float* __restrict__ pw,  const float* __restrict__ pb,
                          const float* __restrict__ plw, const float* __restrict__ plb,
                          float* __restrict__ out)
{
    int row = blockIdx.x;
    int n  = row % N_;
    int bt = row / N_;
    const float* xr = x + (size_t)row * CIN_;
    __shared__ float xs[CIN_];
    __shared__ float ys[96];
    __shared__ float st[2];
    int tid = threadIdx.x;
    if (tid < CIN_) xs[tid] = xr[tid];
    __syncthreads();
    if (tid == 0) {
        float s = 0.f; for (int i = 0; i < CIN_; ++i) s += xs[i];
        float mu = s / CIN_, v = 0.f;
        for (int i = 0; i < CIN_; ++i) { float d = xs[i] - mu; v += d * d; }
        st[0] = mu; st[1] = rsqrtf(v / CIN_ + 1e-5f);
    }
    __syncthreads();
    float mu = st[0], rs = st[1];
    float acc = pb[tid];
    for (int i = 0; i < CIN_; ++i) {
        float xi = (xs[i] - mu) * rs * inw[i] + inb[i];
        acc += xi * pw[tid * CIN_ + i];
    }
    ys[tid] = acc;
    __syncthreads();
    if (tid == 0) {
        float s = 0.f; for (int i = 0; i < 96; ++i) s += ys[i];
        float m2 = s / 96.f, v = 0.f;
        for (int i = 0; i < 96; ++i) { float d = ys[i] - m2; v += d * d; }
        st[0] = m2; st[1] = rsqrtf(v / 96.f + 1e-5f);
    }
    __syncthreads();
    float y = (ys[tid] - st[0]) * st[1] * plw[tid] + plb[tid];
    y = gelu_f(y);
    if (n >= 42) y *= x[((size_t)bt * N_ + n) * CIN_ + 9];   // face mask
    out[(size_t)row * 96 + tid] = y;
}

// ---------------------------------------------------------------------------
// A_eff = (A + tanh(adj_res)*0.3) / max(sum|.|, 1) per row, 3x47x47
// ---------------------------------------------------------------------------
__global__ void k_adj(const float* __restrict__ A, const float* __restrict__ ar,
                      float* __restrict__ aeff)
{
    int id = blockIdx.x;            // k*47+n
    int tid = threadIdx.x;          // 64
    __shared__ float rowv[N_];
    __shared__ float ssum;
    if (tid < N_) rowv[tid] = A[id * N_ + tid] + tanhf(ar[id * N_ + tid]) * 0.3f;
    __syncthreads();
    if (tid == 0) {
        float s = 0.f; for (int m = 0; m < N_; ++m) s += fabsf(rowv[m]);
        ssum = fmaxf(s, 1.f);
    }
    __syncthreads();
    if (tid < N_) aeff[id * N_ + tid] = rowv[tid] / ssum;
}

// ---------------------------------------------------------------------------
// agg[bt][n][k*cin+cc] = dw[k,cc] * sum_m Aeff[k,n,m] * h[bt,m,cc]  (bf16 out)
// ---------------------------------------------------------------------------
__global__ void k_agg(const float* __restrict__ h, const float* __restrict__ aeff,
                      const float* __restrict__ dw, __bf16* __restrict__ agg, int cin)
{
    int bt = blockIdx.x;
    extern __shared__ float sm[];
    float* sx = sm;                    // 47*cin
    float* sA = sm + N_ * cin;         // 3*47*47
    int tid = threadIdx.x;             // 256
    for (int e = tid; e < N_ * cin; e += 256) sx[e] = h[(size_t)bt * N_ * cin + e];
    for (int e = tid; e < 3 * N_ * N_; e += 256) sA[e] = aeff[e];
    __syncthreads();
    int c3 = 3 * cin;
    int tot = N_ * c3;
    for (int e = tid; e < tot; e += 256) {
        int n = e / c3, r = e % c3;
        int k = r / cin, cc = r % cin;
        const float* arow = sA + (k * N_ + n) * N_;
        float s = 0.f;
        for (int m = 0; m < N_; ++m) s += arow[m] * sx[m * cin + cc];
        agg[(size_t)bt * tot + e] = f2bf(s * dw[k * cin + cc]);
    }
}

// ---------------------------------------------------------------------------
// depthwise temporal conv over T=32, layout [b,t,n,c]
// ---------------------------------------------------------------------------
__global__ void k_tconv(const float* __restrict__ in, float* __restrict__ out,
                        const float* __restrict__ w, int co, int tk)
{
    int id = blockIdx.x * blockDim.x + threadIdx.x;
    int total = B_ * N_ * co;
    if (id >= total) return;
    int c = id % co, bn = id / co;
    int n = bn % N_, b = bn / N_;
    int pad = tk / 2;
    for (int t = 0; t < T_; ++t) {
        float s = 0.f;
        for (int j = 0; j < tk; ++j) {
            int tt = t + j - pad;
            if (tt >= 0 && tt < T_)
                s += w[c * tk + j] * in[(((size_t)(b * T_ + tt)) * N_ + n) * co + c];
        }
        out[(((size_t)(b * T_ + t)) * N_ + n) * co + c] = s;
    }
}

// ---------------------------------------------------------------------------
// groupnorm: 8 groups over (co/8, T) per (b,n); in-place
// ---------------------------------------------------------------------------
__global__ void k_gn(float* __restrict__ h, const float* __restrict__ gw,
                     const float* __restrict__ gb, int co)
{
    int id = blockIdx.x;
    int grp = id % 8, bn = id / 8;
    int n = bn % N_, b = bn / N_;
    int cg = co / 8;
    int tid = threadIdx.x;          // 128
    __shared__ float s1[128], s2[128];
    int total = cg * T_;
    float sum = 0.f, sq = 0.f;
    for (int e = tid; e < total; e += 128) {
        int ci = e % cg, t = e / cg;
        float v = h[(((size_t)(b * T_ + t)) * N_ + n) * co + grp * cg + ci];
        sum += v; sq += v * v;
    }
    s1[tid] = sum; s2[tid] = sq;
    __syncthreads();
    for (int off = 64; off > 0; off >>= 1) {
        if (tid < off) { s1[tid] += s1[tid + off]; s2[tid] += s2[tid + off]; }
        __syncthreads();
    }
    float mu = s1[0] / total;
    float var = s2[0] / total - mu * mu;
    float rs = rsqrtf(var + 1e-5f);
    for (int e = tid; e < total; e += 128) {
        int ci = e % cg, t = e / cg;
        int c = grp * cg + ci;
        size_t off = (((size_t)(b * T_ + t)) * N_ + n) * co + c;
        h[off] = (h[off] - mu) * rs * gw[c] + gb[c];
    }
}

// ---------------------------------------------------------------------------
// generic: y = [gelu]( LN(in + res) * w + b ); writes f32 and/or bf16
// ---------------------------------------------------------------------------
__global__ void k_ln(const float* __restrict__ in, const float* __restrict__ res,
                     float* __restrict__ outf, __bf16* __restrict__ outbf,
                     const float* __restrict__ w, const float* __restrict__ b,
                     int dim, int act)
{
    int row = blockIdx.x;
    const float* ir = in + (size_t)row * dim;
    const float* rr = res ? res + (size_t)row * dim : nullptr;
    __shared__ float s1[128], s2[128];
    int tid = threadIdx.x;
    float sum = 0.f, sq = 0.f;
    for (int c = tid; c < dim; c += 128) {
        float v = ir[c] + (rr ? rr[c] : 0.f);
        sum += v; sq += v * v;
    }
    s1[tid] = sum; s2[tid] = sq;
    __syncthreads();
    for (int off = 64; off > 0; off >>= 1) {
        if (tid < off) { s1[tid] += s1[tid + off]; s2[tid] += s2[tid + off]; }
        __syncthreads();
    }
    float mu = s1[0] / dim;
    float var = s2[0] / dim - mu * mu;
    float rs = rsqrtf(var + 1e-5f);
    for (int c = tid; c < dim; c += 128) {
        float v = ir[c] + (rr ? rr[c] : 0.f);
        float y = (v - mu) * rs * w[c] + b[c];
        if (act) y = gelu_f(y);
        if (outf)  outf[(size_t)row * dim + c] = y;
        if (outbf) outbf[(size_t)row * dim + c] = f2bf(y);
    }
}

// ---------------------------------------------------------------------------
// node attention pooling
// ---------------------------------------------------------------------------
__global__ void k_attnpool(const float* __restrict__ a1, const float* __restrict__ w2,
                           const float* __restrict__ b2, const float* __restrict__ h,
                           float* __restrict__ pooled)
{
    int bt = blockIdx.x;
    __shared__ float prob[N_];
    int tid = threadIdx.x;   // 256
    if (tid < N_) {
        const float* ar = a1 + ((size_t)bt * N_ + tid) * 96;
        float s = b2[0];
        for (int c = 0; c < 96; ++c) s += ar[c] * w2[c];
        prob[tid] = s;
    }
    __syncthreads();
    if (tid == 0) {
        float mx = prob[0];
        for (int n = 1; n < N_; ++n) mx = fmaxf(mx, prob[n]);
        float sum = 0.f;
        for (int n = 0; n < N_; ++n) { float e = expf(prob[n] - mx); prob[n] = e; sum += e; }
        float inv = 1.f / sum;
        for (int n = 0; n < N_; ++n) prob[n] *= inv;
    }
    __syncthreads();
    for (int d = tid; d < D_; d += blockDim.x) {
        float s = 0.f;
        for (int n = 0; n < N_; ++n) s += prob[n] * h[((size_t)bt * N_ + n) * D_ + d];
        pooled[(size_t)bt * D_ + d] = s;
    }
}

// ---------------------------------------------------------------------------
// geometric features (76) + their layernorm, one thread per (b,t)
// ---------------------------------------------------------------------------
__global__ void k_geo(const float* __restrict__ x, const float* __restrict__ gw,
                      const float* __restrict__ gb, float* __restrict__ geo)
{
    int bt = blockIdx.x * blockDim.x + threadIdx.x;
    if (bt >= TOK_) return;
    const float* xr = x + (size_t)bt * N_ * CIN_;
    float p[N_][3];
    for (int n = 0; n < N_; ++n)
        for (int j = 0; j < 3; ++j) p[n][j] = xr[n * CIN_ + j];

    auto dist = [&](int a, int b2) {
        float dx = p[a][0]-p[b2][0], dy = p[a][1]-p[b2][1], dz = p[a][2]-p[b2][2];
        return sqrtf(dx*dx + dy*dy + dz*dz + 1e-6f);
    };
    auto nrm3 = [&](float vx, float vy, float vz) { return sqrtf(vx*vx + vy*vy + vz*vz); };

    float f[NGEO_];
    int idx = 0;
    for (int hh = 0; hh < 2; ++hh) {
        int o = hh * 21;
        f[idx++] = dist(o+4,  o+8);  f[idx++] = dist(o+8,  o+12);
        f[idx++] = dist(o+12, o+16); f[idx++] = dist(o+16, o+20);
        f[idx++] = dist(o+4,  o+20);
        f[idx++] = dist(o+2,  o+4)  / (dist(o+2,  o+3)  + 1e-4f);
        f[idx++] = dist(o+5,  o+8)  / (dist(o+5,  o+6)  + 1e-4f);
        f[idx++] = dist(o+9,  o+12) / (dist(o+9,  o+10) + 1e-4f);
        f[idx++] = dist(o+13, o+16) / (dist(o+13, o+14) + 1e-4f);
        f[idx++] = dist(o+17, o+20) / (dist(o+17, o+18) + 1e-4f);
        f[idx++] = p[o+8][0] - p[o+12][0];
        f[idx++] = dist(o+4, o+5);
    }
    float fg = xr[42 * CIN_ + 9];
    const int fp[10][2] = {{0,42},{0,43},{0,44},{21,42},{21,43},{21,44},
                           {8,42},{8,44},{29,42},{29,44}};
    for (int i = 0; i < 10; ++i) f[idx++] = dist(fp[i][0], fp[i][1]) * fg;

    const int TRIP[15][3] = {{0,1,2},{1,2,3},{2,3,4},{0,5,6},{5,6,7},{6,7,8},
                             {0,9,10},{9,10,11},{10,11,12},{0,13,14},{13,14,15},
                             {14,15,16},{0,17,18},{17,18,19},{18,19,20}};
    for (int hh = 0; hh < 2; ++hh) {
        int o = hh * 21;
        for (int i = 0; i < 15; ++i) {
            int a = TRIP[i][0]+o, b2 = TRIP[i][1]+o, c = TRIP[i][2]+o;
            float v1x = p[a][0]-p[b2][0], v1y = p[a][1]-p[b2][1], v1z = p[a][2]-p[b2][2];
            float v2x = p[c][0]-p[b2][0], v2y = p[c][1]-p[b2][1], v2z = p[c][2]-p[b2][2];
            float cs = (v1x*v2x + v1y*v2y + v1z*v2z) /
                       (nrm3(v1x,v1y,v1z) * nrm3(v2x,v2y,v2z) + 1e-6f);
            cs = fminf(fmaxf(cs, -1.f + 1e-6f), 1.f - 1e-6f);
            f[idx++] = acosf(cs);
        }
    }
    for (int hh = 0; hh < 2; ++hh) {
        int o = hh * 21;
        float ax = p[o+5][0]-p[o][0], ay = p[o+5][1]-p[o][1], az = p[o+5][2]-p[o][2];
        float bx = p[o+17][0]-p[o][0], by = p[o+17][1]-p[o][1], bz = p[o+17][2]-p[o][2];
        float cx = ay*bz - az*by, cy = az*bx - ax*bz, cz = ax*by - ay*bx;
        float inv = 1.f / (nrm3(cx,cy,cz) + 1e-6f);
        f[idx++] = cx*inv; f[idx++] = cy*inv; f[idx++] = cz*inv;
    }
    const int SPR[4] = {5, 9, 13, 17};
    for (int hh = 0; hh < 2; ++hh) {
        int o = hh * 21;
        for (int i = 0; i < 3; ++i) {
            float s1x = p[o+SPR[i]][0]-p[o][0], s1y = p[o+SPR[i]][1]-p[o][1], s1z = p[o+SPR[i]][2]-p[o][2];
            float s2x = p[o+SPR[i+1]][0]-p[o][0], s2y = p[o+SPR[i+1]][1]-p[o][1], s2z = p[o+SPR[i+1]][2]-p[o][2];
            float cs = (s1x*s2x + s1y*s2y + s1z*s2z) /
                       (nrm3(s1x,s1y,s1z) * nrm3(s2x,s2y,s2z) + 1e-6f);
            cs = fminf(fmaxf(cs, -1.f + 1e-6f), 1.f - 1e-6f);
            f[idx++] = acosf(cs);
        }
    }
    float s = 0.f;
    for (int i = 0; i < NGEO_; ++i) s += f[i];
    float mu = s / NGEO_, v = 0.f;
    for (int i = 0; i < NGEO_; ++i) { float d = f[i] - mu; v += d * d; }
    float rs = rsqrtf(v / NGEO_ + 1e-5f);
    for (int i = 0; i < NGEO_; ++i)
        geo[(size_t)bt * NGEO_ + i] = (f[i] - mu) * rs * gw[i] + gb[i];
}

// concat [pooled(384) | geo(76) | zeros(20)] -> bf16 [TOK_, 480]
__global__ void k_cat(const float* __restrict__ pooled, const float* __restrict__ geo,
                      __bf16* __restrict__ cat)
{
    int id = blockIdx.x * blockDim.x + threadIdx.x;
    int tot = TOK_ * KGEO_;
    if (id >= tot) return;
    int j = id % KGEO_, bt = id / KGEO_;
    float v = 0.f;
    if (j < D_)               v = pooled[(size_t)bt * D_ + j];
    else if (j < D_ + NGEO_)  v = geo[(size_t)bt * NGEO_ + (j - D_)];
    cat[id] = f2bf(v);
}

__global__ void k_addpos(float* __restrict__ xx, const float* __restrict__ pe)
{
    int id = blockIdx.x * blockDim.x + threadIdx.x;
    if (id >= TOK_ * D_) return;
    int d = id % D_, bt = id / D_;
    xx[id] += pe[(bt % T_) * D_ + d];
}

// ---------------------------------------------------------------------------
// MHA core: per (b, head) 32x32 attention with dh=48; bf16 output
// ---------------------------------------------------------------------------
__global__ void k_attn(const float* __restrict__ qkv, __bf16* __restrict__ out)
{
    int bh = blockIdx.x;
    int b = bh >> 3, hh = bh & 7;
    __shared__ float sQ[T_*48], sK[T_*48], sV[T_*48], sS[T_*T_];
    int tid = threadIdx.x;   // 256
    for (int e = tid; e < T_*48; e += 256) {
        int t = e / 48, d = e % 48;
        size_t base = ((size_t)(b * T_ + t)) * 1152 + hh * 48 + d;
        sQ[e] = qkv[base]; sK[e] = qkv[base + 384]; sV[e] = qkv[base + 768];
    }
    __syncthreads();
    const float scale = 0.14433756729740646f;   // 1/sqrt(48)
    for (int e = tid; e < T_*T_; e += 256) {
        int i = e >> 5, j = e & 31;
        float s = 0.f;
        for (int d = 0; d < 48; ++d) s += sQ[i*48+d] * sK[j*48+d];
        sS[e] = s * scale;
    }
    __syncthreads();
    if (tid < T_) {
        float mx = sS[tid*T_];
        for (int j = 1; j < T_; ++j) mx = fmaxf(mx, sS[tid*T_+j]);
        float sum = 0.f;
        for (int j = 0; j < T_; ++j) { float e = expf(sS[tid*T_+j]-mx); sS[tid*T_+j] = e; sum += e; }
        float inv = 1.f / sum;
        for (int j = 0; j < T_; ++j) sS[tid*T_+j] *= inv;
    }
    __syncthreads();
    for (int e = tid; e < T_*48; e += 256) {
        int t = e / 48, d = e % 48;
        float s = 0.f;
        for (int j = 0; j < T_; ++j) s += sS[t*T_+j] * sV[j*48+d];
        out[((size_t)(b * T_ + t)) * D_ + hh * 48 + d] = f2bf(s);
    }
}

// ---------------------------------------------------------------------------
// param flatten order (jax tree: nested dicts sorted by key, None dropped)
// ---------------------------------------------------------------------------
enum {
    P_ATTN1_B = 0, P_ATTN1_W, P_ATTN2_B, P_ATTN2_W,
    P_G1_ADJ, P_G1_DW, P_G1_GNB, P_G1_GNW, P_G1_LNB, P_G1_LNW, P_G1_PW, P_G1_RES, P_G1_TW,
    P_G2_ADJ, P_G2_DW, P_G2_GNB, P_G2_GNW, P_G2_LNB, P_G2_LNW, P_G2_PW, P_G2_TW,
    P_G3_ADJ, P_G3_DW, P_G3_GNB, P_G3_GNW, P_G3_LNB, P_G3_LNW, P_G3_PW, P_G3_RES, P_G3_TW,
    P_GEO_NB, P_GEO_NW, P_GEO_PB, P_GEO_PW, P_IN_NB, P_IN_NW,
    P_L0 = 36,           // per layer: ff1_b, ff1_w, ff2_b, ff2_w, in_b, in_w,
                         //            ln1_b, ln1_w, ln2_b, ln2_w, out_b, out_w
    P_POS = 36 + 6*12, P_PROJ_B, P_PROJ_LNB, P_PROJ_LNW, P_PROJ_W, P_TN_B, P_TN_W
};

extern "C" void kernel_launch(void* const* d_in, const int* in_sizes, int n_in,
                              void* d_out, int out_size, void* d_ws, size_t ws_size,
                              hipStream_t stream)
{
    (void)out_size; (void)ws_size;
    const float *xin, *Aadj;
    int pbase = 1;
    if (in_sizes[0] == B_*T_*N_*CIN_) { xin = (const float*)d_in[0]; Aadj = (const float*)d_in[n_in-1]; }
    else                              { Aadj = (const float*)d_in[0]; xin = (const float*)d_in[n_in-1]; }
    auto P = [&](int i) { return (const float*)d_in[pbase + i]; };

    char* base = (char*)d_ws;
    size_t o = 0;
    auto carveB = [&](size_t bytes) { char* p = base + o; o += (bytes + 255) & ~(size_t)255; return p; };
    auto carveF = [&](size_t nf) { return (float*)carveB(nf * 4); };
    auto carveH = [&](size_t nh) { return (__bf16*)carveB(nh * 2); };

    float*  hA    = carveF((size_t)ROWS_ * D_);
    float*  hD    = carveF((size_t)ROWS_ * D_);
    float*  hB    = carveF((size_t)ROWS_ * D_);
    float*  hC    = carveF((size_t)ROWS_ * D_);
    float*  resb  = carveF((size_t)ROWS_ * D_);
    float*  aeff  = carveF(3 * N_ * N_);
    float*  a1    = carveF((size_t)ROWS_ * 96);
    float*  pooled= carveF((size_t)TOK_ * D_);
    float*  geo   = carveF((size_t)TOK_ * NGEO_);
    float*  tok   = carveF((size_t)TOK_ * D_);
    float*  qkvb  = carveF((size_t)TOK_ * 3 * D_);
    __bf16* aggbf = carveH((size_t)ROWS_ * 576);
    __bf16* xibf  = carveH((size_t)ROWS_ * D_);
    __bf16* wbf   = carveH((size_t)1536 * D_);
    __bf16* lnbbf = carveH((size_t)TOK_ * D_);
    __bf16* attbf = carveH((size_t)TOK_ * D_);
    __bf16* ffbf  = carveH((size_t)TOK_ * 4 * D_);
    __bf16* catbf = carveH((size_t)TOK_ * KGEO_);

    auto tobf = [&](const float* in, __bf16* out2, size_t n) {
        long n4 = (long)(n / 4);
        k_tobf<<<(int)((n4 + 255) / 256), 256, 0, stream>>>(in, out2, n4);
    };
    auto gemm = [&](const __bf16* A, const __bf16* W, const float* bias,
                    const float* res, float* C, __bf16* Cbf, int M, int N, int K, int act) {
        int ngroups = (M / 64) * (N / 16);
        k_gemm<<<(ngroups + 3) / 4, 128, 0, stream>>>(A, W, bias, res, C, Cbf, M, N, K, act);
    };

    // input projection
    k_in_proj<<<ROWS_, 96, 0, stream>>>(xin, P(P_IN_NW), P(P_IN_NB),
                                        P(P_PROJ_W), P(P_PROJ_B),
                                        P(P_PROJ_LNW), P(P_PROJ_LNB), hA);

    auto gcn = [&](const float* xi, float* xo, int cin, int cout, int tk,
                   const float* adjres, const float* dw,
                   const float* gnw, const float* gnb,
                   const float* lnw, const float* lnbp,
                   const float* pw, const float* resw, const float* tw) {
        k_adj<<<3 * N_, 64, 0, stream>>>(Aadj, adjres, aeff);
        size_t sh = (size_t)(N_ * cin + 3 * N_ * N_) * sizeof(float);
        k_agg<<<TOK_, 256, sh, stream>>>(xi, aeff, dw, aggbf, cin);
        tobf(pw, wbf, (size_t)cout * 3 * cin);
        gemm(aggbf, wbf, nullptr, nullptr, hB, nullptr, ROWS_, cout, 3 * cin, 0);
        const float* rptr = xi;
        if (resw) {
            tobf(resw, wbf, (size_t)cout * cin);
            tobf(xi, xibf, (size_t)ROWS_ * cin);
            gemm(xibf, wbf, nullptr, nullptr, resb, nullptr, ROWS_, cout, cin, 0);
            rptr = resb;
        }
        int tot = B_ * N_ * cout;
        k_tconv<<<(tot + 255) / 256, 256, 0, stream>>>(hB, hC, tw, cout, tk);
        k_gn<<<B_ * N_ * 8, 128, 0, stream>>>(hC, gnw, gnb, cout);
        k_ln<<<ROWS_, 128, 0, stream>>>(hC, rptr, xo, nullptr, lnw, lnbp, cout, 1);
    };

    gcn(hA, hD,  96, 192, 3, P(P_G1_ADJ), P(P_G1_DW), P(P_G1_GNW), P(P_G1_GNB),
        P(P_G1_LNW), P(P_G1_LNB), P(P_G1_PW), P(P_G1_RES), P(P_G1_TW));
    gcn(hD, hD, 192, 192, 3, P(P_G2_ADJ), P(P_G2_DW), P(P_G2_GNW), P(P_G2_GNB),
        P(P_G2_LNW), P(P_G2_LNB), P(P_G2_PW), nullptr,   P(P_G2_TW));
    gcn(hD, hA, 192, 384, 5, P(P_G3_ADJ), P(P_G3_DW), P(P_G3_GNW), P(P_G3_GNB),
        P(P_G3_LNW), P(P_G3_LNB), P(P_G3_PW), P(P_G3_RES), P(P_G3_TW));

    // node attention pooling
    tobf(P(P_ATTN1_W), wbf, (size_t)96 * D_);
    tobf(hA, xibf, (size_t)ROWS_ * D_);
    gemm(xibf, wbf, P(P_ATTN1_B), nullptr, a1, nullptr, ROWS_, 96, D_, 1);
    k_attnpool<<<TOK_, 256, 0, stream>>>(a1, P(P_ATTN2_W), P(P_ATTN2_B), hA, pooled);

    // geometric features (includes geo layernorm)
    k_geo<<<(TOK_ + 127) / 128, 128, 0, stream>>>(xin, P(P_GEO_NW), P(P_GEO_NB), geo);

    // concat (bf16, zero-padded to K=480) + geo projection + pos enc
    {
        int tot = TOK_ * KGEO_;
        k_cat<<<(tot + 255) / 256, 256, 0, stream>>>(pooled, geo, catbf);
    }
    k_padw<<<(D_ * KGEO_ + 255) / 256, 256, 0, stream>>>(P(P_GEO_PW), wbf);
    gemm(catbf, wbf, P(P_GEO_PB), nullptr, tok, nullptr, TOK_, D_, KGEO_, 0);
    {
        int tot = TOK_ * D_;
        k_addpos<<<(tot + 255) / 256, 256, 0, stream>>>(tok, P(P_POS));
    }

    // transformer layers
    for (int l = 0; l < 6; ++l) {
        int pb2 = P_L0 + 12 * l;
        const float *ff1_b = P(pb2+0), *ff1_w = P(pb2+1),
                    *ff2_b = P(pb2+2), *ff2_w = P(pb2+3),
                    *in_b  = P(pb2+4), *in_w  = P(pb2+5),
                    *ln1_b = P(pb2+6), *ln1_w = P(pb2+7),
                    *ln2_b = P(pb2+8), *ln2_w = P(pb2+9),
                    *out_b = P(pb2+10), *out_w = P(pb2+11);
        k_ln<<<TOK_, 128, 0, stream>>>(tok, nullptr, nullptr, lnbbf, ln1_w, ln1_b, D_, 0);
        tobf(in_w, wbf, (size_t)3 * D_ * D_);
        gemm(lnbbf, wbf, in_b, nullptr, qkvb, nullptr, TOK_, 3 * D_, D_, 0);
        k_attn<<<B_ * 8, 256, 0, stream>>>(qkvb, attbf);
        tobf(out_w, wbf, (size_t)D_ * D_);
        gemm(attbf, wbf, out_b, tok, tok, nullptr, TOK_, D_, D_, 0);          // + residual
        k_ln<<<TOK_, 128, 0, stream>>>(tok, nullptr, nullptr, lnbbf, ln2_w, ln2_b, D_, 0);
        tobf(ff1_w, wbf, (size_t)4 * D_ * D_);
        gemm(lnbbf, wbf, ff1_b, nullptr, nullptr, ffbf, TOK_, 4 * D_, D_, 1); // gelu -> bf16
        tobf(ff2_w, wbf, (size_t)D_ * 4 * D_);
        gemm(ffbf, wbf, ff2_b, tok, tok, nullptr, TOK_, D_, 4 * D_, 0);       // + residual
    }

    // final layernorm -> output
    k_ln<<<TOK_, 128, 0, stream>>>(tok, nullptr, (float*)d_out, nullptr,
                                   P(P_TN_W), P(P_TN_B), D_, 0);
}
